// KWconvNd_63857573757044
// MI455X (gfx1250) — matrix operand order repaired and checked by reading, not compiled, not run on hardware
//
#include <hip/hip_runtime.h>
#include <hip/hip_bf16.h>

// ---------------------------------------------------------------------------
// KWConv2d for MI455X (gfx1250): attention -> WMMA weight-aggregation GEMM ->
// per-sample implicit-GEMM conv on v_wmma_f32_16x16x32_f16, double-buffered
// LDS, K axis reordered as k' = (kh*3+kw)*256 + ci so each 32-wide K tile has
// fixed (kh,kw) and a contiguous channel slice (no div/mod in the hot loop).
// A-staging uses gfx1250 async global->LDS DMA (ASYNCcnt-tracked).
// ---------------------------------------------------------------------------

typedef __attribute__((ext_vector_type(16))) _Float16 v16h;
typedef __attribute__((ext_vector_type(8)))  _Float16 v8h;
typedef __attribute__((ext_vector_type(4)))  _Float16 v4h;
typedef __attribute__((ext_vector_type(8)))  float    v8f;
typedef __attribute__((ext_vector_type(4)))  int      v4i;

#define BATCH     32
#define CIN       256
#define HW        3136      // 56*56
#define HH        56
#define OUTC      256
#define NCELLS    32
#define HIDDEN    16
#define KWP       528       // 33 * 16
#define PMIX      33
#define MMIX      16
#define KIM       2304      // CIN * 9
#define WPER_B    589824    // OUTC * KIM
#define CELLROW   36864     // 64*64*9
#define LDST      40        // LDS row stride in halfs (80 B, 16B aligned, conflict-free)

#if defined(__has_builtin)
#if __has_builtin(__builtin_amdgcn_global_load_async_to_lds_b128) && \
    __has_builtin(__builtin_amdgcn_s_wait_asynccnt)
#define USE_ASYNC_LDS 1
#endif
#endif
#ifndef USE_ASYNC_LDS
#define USE_ASYNC_LDS 0
#endif

#if USE_ASYNC_LDS
// builtin signature (from compiler diagnostic): (v4i AS1*, v4i AS3*, Ii, Ii)
#define GAS1(p) ((__attribute__((address_space(1))) v4i*)(p))
#define LAS3(p) ((__attribute__((address_space(3))) v4i*)(p))
#endif

// ---------------------------------------------------------------------------
// Kernel 1: attention.  One block per batch sample.  Produces att[512][32] f32.
// ---------------------------------------------------------------------------
__global__ __launch_bounds__(256) void kw_attention(
    const float* __restrict__ x,
    const float* __restrict__ fc1_w, const float* __restrict__ fc1_b,
    const float* __restrict__ ln_g,  const float* __restrict__ ln_b,
    const float* __restrict__ sp_w,  const float* __restrict__ sp_b,
    float* __restrict__ att)
{
  __shared__ float pooled[CIN];
  __shared__ float hbuf[HIDDEN];
  __shared__ float abuf[KWP];

  const int b    = blockIdx.x;
  const int t    = threadIdx.x;
  const int wave = t >> 5;
  const int lane = t & 31;
  const float* xb = x + (size_t)b * CIN * HW;

  for (int c0 = 0; c0 < CIN; c0 += 8) {
    const int c = c0 + wave;
    const float* xc = xb + (size_t)c * HW;
    float s = 0.f;
    for (int i = lane; i < HW; i += 32) s += xc[i];
    #pragma unroll
    for (int off = 16; off > 0; off >>= 1) s += __shfl_down(s, off, 32);
    if (lane == 0) pooled[c] = s * (1.0f / (float)HW);
  }
  __syncthreads();

  if (t < HIDDEN) {
    float h = fc1_b[t];
    for (int c = 0; c < CIN; ++c) h += pooled[c] * fc1_w[t * CIN + c];
    hbuf[t] = h;
  }
  __syncthreads();

  if (t == 0) {
    float mu = 0.f;
    for (int i = 0; i < HIDDEN; ++i) mu += hbuf[i];
    mu *= (1.0f / HIDDEN);
    float var = 0.f;
    for (int i = 0; i < HIDDEN; ++i) { float d = hbuf[i] - mu; var += d * d; }
    var *= (1.0f / HIDDEN);
    float inv = rsqrtf(var + 1e-5f);
    for (int i = 0; i < HIDDEN; ++i)
      hbuf[i] = fmaxf((hbuf[i] - mu) * inv * ln_g[i] + ln_b[i], 0.0f);
  }
  __syncthreads();

  for (int j = t; j < KWP; j += 256) {
    float a = sp_b[j];
    #pragma unroll
    for (int k = 0; k < HIDDEN; ++k) a += hbuf[k] * sp_w[j * HIDDEN + k];
    abuf[j] = a;
  }
  __syncthreads();

  if (t < MMIX) {
    float s = 0.001f;
    for (int p = 0; p < PMIX; ++p) s += fabsf(abuf[t * PMIX + p]);
    const float inv = 1.0f / s;
    float* dst = att + ((size_t)b * MMIX + t) * NCELLS;
    for (int cell = 0; cell < NCELLS; ++cell)
      dst[cell] = abuf[t * PMIX + cell] * inv;
  }
}

// ---------------------------------------------------------------------------
// Kernel 2: weight aggregation GEMM  agg = att[512,32] @ cells[32,36864]
// via v_wmma_f32_16x16x32_f16, scatter-stored f16 in the permuted, K-reordered
// layout:  Wf16[b][o][ k' = q*256 + (gi*64+ci) ]  where o = go*64+co.
// ---------------------------------------------------------------------------
__global__ __launch_bounds__(256) void kw_aggregate(
    const float* __restrict__ att,
    const float* __restrict__ cells,
    _Float16* __restrict__ Wf16)
{
  const int wave = threadIdx.x >> 5;
  const int lane = threadIdx.x & 31;
  const int job  = blockIdx.x * 8 + wave;   // 1152 blocks * 8 = 9216
  const int rt    = job / 288;              // row tile 0..31
  const int chunk = job % 288;              // 8 col tiles each -> 2304 total
  const int g    = lane >> 4;
  const int ln16 = lane & 15;

  // A fragment: rows rt*16..+15 of att, K=0..31, converted f32->f16.
  const float* ar = att + (size_t)(rt * 16 + ln16) * NCELLS;
  v16h afrag;
  #pragma unroll
  for (int h = 0; h < 8; ++h) afrag[h] = (_Float16)ar[16 * g + h];
  #pragma unroll
  for (int h = 0; h < 8; ++h) afrag[8 + h] = (_Float16)ar[16 * g + 8 + h];

  for (int ct8 = 0; ct8 < 8; ++ct8) {
    const int ct = chunk * 8 + ct8;
    const int j  = ct * 16 + ln16;          // column in [0, 36864)

    v16h bfrag;
    #pragma unroll
    for (int h = 0; h < 16; ++h)
      bfrag[h] = (_Float16)cells[(size_t)(16 * g + h) * CELLROW + j];

    v8f c = {};
    c = __builtin_amdgcn_wmma_f32_16x16x32_f16(
        false, afrag, false, bfrag, (short)0, c, false, false);

    // j = co*576 + rem, rem = ci*9 + q  ->  k' = q*256 + gi*64 + ci
    const int co  = j / 576;
    const int rem = j - co * 576;
    const int ci  = rem / 9;
    const int q   = rem - ci * 9;
    #pragma unroll
    for (int r = 0; r < 8; ++r) {
      const int mrow = rt * 16 + 8 * g + r;   // agg row = b*16 + (go*4+gi)
      const int bb = mrow >> 4;
      const int mm = mrow & 15;
      const int go = mm >> 2;
      const int gi = mm & 3;
      const size_t dst = (size_t)bb * WPER_B
                       + (size_t)(go * 64 + co) * KIM
                       + (size_t)(q * 256 + gi * 64 + ci);
      Wf16[dst] = (_Float16)c[r];
    }
  }
}

// ---------------------------------------------------------------------------
// Kernel 3: per-sample implicit-GEMM conv, double-buffered LDS.
//   per b:  C[256, 3136] = Wf16_b[256, 2304] x im2col(x_b)[2304, 3136]
// K tile kt: q = kt/8 fixed (kh,kw), channels ci0 = (kt%8)*32 .. +31.
// Block tile 128(M) x 128(N), 8 waves (2M x 4N), wave tile 64x32 -> 8 WMMA/iter.
// ---------------------------------------------------------------------------
__global__ __launch_bounds__(256) void kw_conv(
    const float* __restrict__ x,
    const _Float16* __restrict__ Wf16,
    float* __restrict__ out)
{
  __shared__ __attribute__((aligned(16))) _Float16 Alds[2][128 * LDST]; // [m][k]
  __shared__ __attribute__((aligned(16))) _Float16 Blds[2][128 * LDST]; // [n][k]

  const int b  = blockIdx.z;
  const int m0 = blockIdx.y * 128;
  const int n0 = blockIdx.x * 128;
  const int t    = threadIdx.x;
  const int wave = t >> 5;
  const int lane = t & 31;
  const int wm   = wave & 1;
  const int wn   = wave >> 1;
  const int g    = lane >> 4;
  const int ln16 = lane & 15;

  const float*    xb = x    + (size_t)b * CIN * HW;
  const _Float16* Wb = Wf16 + (size_t)b * WPER_B + (size_t)m0 * KIM;

  // per-thread im2col invariants: one spatial position, 16 channel slots
  const int nl = t & 127;                 // local n
  const int kb = t >> 7;                  // channel slot parity (0/1)
  const int ng = n0 + nl;
  const bool nvalid = ng < HW;
  const int y  = ng / HH;
  const int xx = ng - y * HH;

  float breg[16];
#if !USE_ASYNC_LDS
  v4h   areg[4];
#endif

  // --- staging helpers -----------------------------------------------------
  auto prefB = [&](int kt) {
    const int q   = kt >> 3;
    const int ci0 = (kt & 7) * 32;
    const int kh  = q / 3;
    const int kw  = q - kh * 3;
    const int iy  = y + kh - 1;
    const int ix  = xx + kw - 1;
    const bool v  = nvalid && (unsigned)iy < (unsigned)HH
                           && (unsigned)ix < (unsigned)HH;
    if (v) {
      const float* p = xb + (size_t)(ci0 + kb) * HW + (iy * HH + ix);
      #pragma unroll
      for (int s = 0; s < 16; ++s) breg[s] = p[(size_t)(2 * s) * HW];
    } else {
      #pragma unroll
      for (int s = 0; s < 16; ++s) breg[s] = 0.0f;
    }
  };
  auto writeB = [&](int c) {
    #pragma unroll
    for (int s = 0; s < 16; ++s)
      Blds[c][nl * LDST + kb + 2 * s] = (_Float16)breg[s];
  };
#if USE_ASYNC_LDS
  auto stageA = [&](int kt, int c) {            // async DMA, 2 b128 per lane
    const int k0 = kt * 32;
    #pragma unroll
    for (int s = 0; s < 2; ++s) {
      const int ch = t * 2 + s;                 // 0..511 chunks of 16B
      const int m  = ch >> 2;
      const int c4 = ch & 3;
      __builtin_amdgcn_global_load_async_to_lds_b128(
          GAS1(Wb + (size_t)m * KIM + k0 + c4 * 8),
          LAS3(&Alds[c][m * LDST + c4 * 8]), 0, 0);
    }
  };
#else
  auto prefA = [&](int kt) {
    const int k0 = kt * 32;
    #pragma unroll
    for (int s = 0; s < 4; ++s) {
      const int j = t + 256 * s;
      const int m = j >> 3;
      const int k4 = (j & 7) << 2;
      areg[s] = *(const v4h*)&Wb[(size_t)m * KIM + k0 + k4];
    }
  };
  auto writeA = [&](int c) {
    #pragma unroll
    for (int s = 0; s < 4; ++s) {
      const int j = t + 256 * s;
      const int m = j >> 3;
      const int k4 = (j & 7) << 2;
      *(v4h*)&Alds[c][m * LDST + k4] = areg[s];
    }
  };
#endif

  v8f acc[4][2] = {};
  int cur = 0;

#if USE_ASYNC_LDS
  stageA(0, 0);
#else
  prefA(0);
#endif
  prefB(0);

  for (int kt = 0; kt < KIM / 32; ++kt) {
#if !USE_ASYNC_LDS
    writeA(cur);
#endif
    writeB(cur);
#if USE_ASYNC_LDS
    __builtin_amdgcn_s_wait_asynccnt(0);
#endif
    __syncthreads();

    if (kt < KIM / 32 - 1) {               // prefetch next tile
#if USE_ASYNC_LDS
      stageA(kt + 1, cur ^ 1);
#else
      prefA(kt + 1);
#endif
      prefB(kt + 1);
    }

    // fragments (ds_load_b128 pairs, conflict-free with 40-half stride)
    v16h af[4], bf[2];
    #pragma unroll
    for (int mf = 0; mf < 4; ++mf) {
      const int row = wm * 64 + mf * 16 + ln16;
      v8h lo = *(const v8h*)&Alds[cur][row * LDST + 16 * g];
      v8h hi = *(const v8h*)&Alds[cur][row * LDST + 16 * g + 8];
      af[mf] = __builtin_shufflevector(lo, hi,
               0,1,2,3,4,5,6,7,8,9,10,11,12,13,14,15);
    }
    #pragma unroll
    for (int nf = 0; nf < 2; ++nf) {
      const int nrow = wn * 32 + nf * 16 + ln16;
      v8h lo = *(const v8h*)&Blds[cur][nrow * LDST + 16 * g];
      v8h hi = *(const v8h*)&Blds[cur][nrow * LDST + 16 * g + 8];
      bf[nf] = __builtin_shufflevector(lo, hi,
               0,1,2,3,4,5,6,7,8,9,10,11,12,13,14,15);
    }

    #pragma unroll
    for (int mf = 0; mf < 4; ++mf)
      #pragma unroll
      for (int nf = 0; nf < 2; ++nf)
        acc[mf][nf] = __builtin_amdgcn_wmma_f32_16x16x32_f16(
            false, af[mf], false, bf[nf], (short)0, acc[mf][nf], false, false);

    cur ^= 1;
  }

  // store: C frag -> lane holds N = l%16, rows M = r + 8*(l/16)
  float* ob = out + ((size_t)b * OUTC + m0) * HW;
  #pragma unroll
  for (int mf = 0; mf < 4; ++mf) {
    #pragma unroll
    for (int nf = 0; nf < 2; ++nf) {
      const int n = n0 + wn * 32 + nf * 16 + ln16;
      if (n < HW) {
        const int mbase = wm * 64 + mf * 16 + 8 * g;
        #pragma unroll
        for (int r = 0; r < 8; ++r)
          ob[(size_t)(mbase + r) * HW + n] = acc[mf][nf][r];
      }
    }
  }
}

// ---------------------------------------------------------------------------
// launch
// ---------------------------------------------------------------------------
extern "C" void kernel_launch(void* const* d_in, const int* in_sizes, int n_in,
                              void* d_out, int out_size, void* d_ws, size_t ws_size,
                              hipStream_t stream) {
  const float* x     = (const float*)d_in[0];
  const float* cells = (const float*)d_in[1];
  const float* fc1_w = (const float*)d_in[2];
  const float* fc1_b = (const float*)d_in[3];
  const float* ln_g  = (const float*)d_in[4];
  const float* ln_b  = (const float*)d_in[5];
  const float* sp_w  = (const float*)d_in[6];
  const float* sp_b  = (const float*)d_in[7];
  float* out = (float*)d_out;

  float*    att  = (float*)d_ws;
  _Float16* Wf16 = (_Float16*)((char*)d_ws + 65536);

  kw_attention<<<BATCH, 256, 0, stream>>>(x, fc1_w, fc1_b, ln_g, ln_b,
                                          sp_w, sp_b, att);

  kw_aggregate<<<1152, 256, 0, stream>>>(att, cells, Wf16);

  dim3 grid((HW + 127) / 128, OUTC / 128, BATCH);   // 25 x 2 x 32
  kw_conv<<<grid, 256, 0, stream>>>(x, Wf16, out);
}
